// MultiheadAttentionBlock_36807869727317
// MI455X (gfx1250) — matrix operand order repaired and checked by reading, not compiled, not run on hardware
//
#include <hip/hip_runtime.h>

// ---------------------------------------------------------------------------
// MHA block for MI455X (gfx1250, wave32, WMMA bf16 16x16x32, f32 accum)
// B=4, S=SK=2048, H=512, NH=8, DS=64
// TDM (tensor_load_to_lds) stages wave-shared tiles, double-buffered with
// compile-time ping-pong; B fragments hoisted ahead of the WMMA chains.
// ---------------------------------------------------------------------------

#define BB   4
#define SS   2048
#define SKK  2048
#define HH   512
#define NHH  8
#define DSS  64
#define MQ   (BB*SS)        // 8192 rows for all GEMMs
#define NPB  (SS*HH)        // 1048576 elements per batch (set-norm domain)
#define EPSL 1e-5f

typedef __bf16 bf16;
typedef __attribute__((ext_vector_type(16))) __bf16 v16bf;
typedef __attribute__((ext_vector_type(8)))  __bf16 v8bf;
typedef __attribute__((ext_vector_type(8)))  float  v8f;
typedef __attribute__((ext_vector_type(4)))  unsigned int u32x4;
typedef __attribute__((ext_vector_type(8)))  int i32x8;
typedef __attribute__((ext_vector_type(4)))  int i32x4;

static __device__ __forceinline__ v16bf combine16(v8bf lo, v8bf hi) {
  v16bf r;
#pragma unroll
  for (int i = 0; i < 8; ++i) { r[i] = lo[i]; r[i + 8] = hi[i]; }
  return r;
}

// ---------------------------------------------------------------------------
// TDM: issue a 2D bf16 tile load Global -> LDS (one issue per wave).
// Descriptor layout per CDNA5 ISA §8.3 (group0) / §8.4 (group1).
// ---------------------------------------------------------------------------
static __device__ __forceinline__ void tdm_load_tile_bf16(
    unsigned lds_off, const void* gptr, unsigned tile_w, unsigned tile_h,
    unsigned stride) {
  const unsigned long long ga = (unsigned long long)gptr;
  u32x4 g0;
  g0[0] = 1u;                                   // count=1, user mode, no gather
  g0[1] = lds_off;                              // lds_addr
  g0[2] = (unsigned)(ga & 0xffffffffu);         // global_addr[31:0]
  g0[3] = (unsigned)((ga >> 32) & 0x01ffffffu)  // global_addr[56:32]
        | (2u << 30);                           // type = 2 ("image")
  i32x8 g1;
  g1[0] = (int)(1u << 16);                      // data_size=1 (2B), mask=0
  g1[1] = (int)((tile_w & 0xffffu) << 16);      // tensor_dim0[15:0] @bit48
  g1[2] = (int)((tile_w >> 16) | ((tile_h & 0xffffu) << 16));  // td0 hi|td1 lo
  g1[3] = (int)((tile_h >> 16) | ((tile_w & 0xffffu) << 16));  // td1 hi|tile0
  g1[4] = (int)(tile_h & 0xffffu);              // tile_dim1, tile_dim2=0
  g1[5] = (int)stride;                          // tensor_dim0_stride[31:0]
  g1[6] = 0;                                    // stride hi | dim1_stride lo
  g1[7] = 0;
  i32x4 z4 = {0, 0, 0, 0};
#if defined(__clang_major__) && (__clang_major__ >= 23)
  i32x8 z8 = {0, 0, 0, 0, 0, 0, 0, 0};
  __builtin_amdgcn_tensor_load_to_lds(g0, g1, z4, z4, z8, 0);
#else
  __builtin_amdgcn_tensor_load_to_lds(g0, g1, z4, z4, 0);
#endif
}

static __device__ __forceinline__ unsigned lds_offset_of(const void* p) {
  return (unsigned)(unsigned long long)p;       // low 32 bits = LDS address
}

// ---------------------------------------------------------------------------
// Deterministic two-stage set-norm statistics (sum, sumsq) per batch
// ---------------------------------------------------------------------------
__global__ void stats_partial_kernel(const float* __restrict__ x,
                                     float* __restrict__ part, int nper) {
  const int b = blockIdx.y;
  const float* xb = x + (size_t)b * nper;
  float s = 0.f, s2 = 0.f;
  for (int i = blockIdx.x * blockDim.x + threadIdx.x; i < nper;
       i += gridDim.x * blockDim.x) {
    float v = xb[i];
    s += v; s2 += v * v;
  }
  __shared__ float sh[256];
  __shared__ float sh2[256];
  const int t = threadIdx.x;
  sh[t] = s; sh2[t] = s2;
  __syncthreads();
  for (int o = 128; o > 0; o >>= 1) {
    if (t < o) { sh[t] += sh[t + o]; sh2[t] += sh2[t + o]; }
    __syncthreads();
  }
  if (t == 0) {
    part[(b * 64 + blockIdx.x) * 2 + 0] = sh[0];
    part[(b * 64 + blockIdx.x) * 2 + 1] = sh2[0];
  }
}

__global__ void stats_final_kernel(const float* __restrict__ part,
                                   float* __restrict__ stats) {
  const int b = blockIdx.x;
  const int t = threadIdx.x;           // 64 threads
  __shared__ float sh[64];
  __shared__ float sh2[64];
  sh[t]  = part[(b * 64 + t) * 2 + 0];
  sh2[t] = part[(b * 64 + t) * 2 + 1];
  __syncthreads();
  for (int o = 32; o > 0; o >>= 1) {
    if (t < o) { sh[t] += sh[t + o]; sh2[t] += sh2[t + o]; }
    __syncthreads();
  }
  if (t == 0) { stats[2 * b] = sh[0]; stats[2 * b + 1] = sh2[0]; }
}

// ---------------------------------------------------------------------------
// set_norm + per-feature affine + convert to bf16
// ---------------------------------------------------------------------------
__global__ void norm_cvt_kernel(const float* __restrict__ x,
                                const float* __restrict__ stats,
                                const float* __restrict__ g,
                                const float* __restrict__ beta,
                                bf16* __restrict__ out, int nper) {
  const size_t tot = (size_t)BB * (size_t)nper;
  for (size_t i = (size_t)blockIdx.x * blockDim.x + threadIdx.x; i < tot;
       i += (size_t)gridDim.x * blockDim.x) {
    const int b = (int)(i / (size_t)nper);
    const int f = (int)(i % (size_t)HH);
    const float inv_n = 1.0f / (float)nper;
    const float mean  = stats[2 * b] * inv_n;
    const float var   = stats[2 * b + 1] * inv_n - mean * mean;
    const float rstd  = rsqrtf(var + EPSL);
    out[i] = (bf16)(((x[i] - mean) * rstd) * g[f] + beta[f]);
  }
}

__global__ void cvt_bf16_kernel(const float* __restrict__ w,
                                bf16* __restrict__ out, int n) {
  for (int i = blockIdx.x * blockDim.x + threadIdx.x; i < n;
       i += gridDim.x * blockDim.x)
    out[i] = (bf16)w[i];
}

// ---------------------------------------------------------------------------
// WMMA GEMM: C[M,N] = A[M,K](bf16,row-major) x W[N,K](bf16,row-major)^T + bias
// mode 0: outb (bf16) = result ; mode 1: outf (f32) = resid + relu(result)
// Block: 256 threads = 8 waves; tile 128(M) x 64(N); wave -> 16x64 stripe.
// Weight k-slab staged by TDM (double-buffered, static ping-pong); A frags
// double-buffered in registers; B frags hoisted ahead of the WMMA chain.
// ---------------------------------------------------------------------------
__global__ __launch_bounds__(256) void gemm_wmma_kernel(
    const bf16* __restrict__ A, const bf16* __restrict__ W,
    const float* __restrict__ bias, bf16* __restrict__ outb,
    const float* __restrict__ resid, float* __restrict__ outf,
    int Mtot, int N, int K, int mode) {
  __shared__ bf16 Wt[2][64 * 32];       // [buf][n_local][k_local]

  const int lane  = threadIdx.x & 31;
  const int wave  = threadIdx.x >> 5;
  const int hl    = lane >> 4;          // 16-lane half
  const int l16   = lane & 15;
  const int selA  = hl ? 16 : 0;        // contiguous-k A-fragment select (k = 16*half + s)
  const int tileM = blockIdx.x * 128 + wave * 16;
  const int tileN = blockIdx.y * 64;
  const bool tdmw = (wave == 0);
  const size_t rowA = (size_t)(tileM + l16) * K;

  const int NK = K / 32;                // 16 (even)
  if (tdmw)
    tdm_load_tile_bf16(lds_offset_of(&Wt[0][0]),
                       W + (size_t)tileN * K, 32, 64, (unsigned)K);
  v8bf alo[2], ahi[2];
  alo[0] = *(const v8bf*)(A + rowA + selA);
  ahi[0] = *(const v8bf*)(A + rowA + selA + 8);

  v8f acc[4] = {};
  auto kstep = [&](int ki, int cur, int nxt) {
    if (tdmw) __builtin_amdgcn_s_wait_tensorcnt(0);
    __syncthreads();                    // Wt[cur] ready; old readers done
    if (ki + 1 < NK) {
      if (tdmw)
        tdm_load_tile_bf16(lds_offset_of(&Wt[nxt][0]),
                           W + (size_t)tileN * K + (ki + 1) * 32, 32, 64,
                           (unsigned)K);
      const size_t ka = rowA + (size_t)(ki + 1) * 32;
      alo[nxt] = *(const v8bf*)(A + ka + selA);
      ahi[nxt] = *(const v8bf*)(A + ka + selA + 8);
    }
    v16bf afrag = combine16(alo[cur], ahi[cur]);
    v16bf bfrag[4];
#pragma unroll
    for (int nt = 0; nt < 4; ++nt) {    // all LDS loads before the WMMA chain
      const bf16* wp = &Wt[cur][(nt * 16 + l16) * 32 + hl * 16];
      bfrag[nt] = combine16(*(const v8bf*)wp, *(const v8bf*)(wp + 8));
    }
#pragma unroll
    for (int nt = 0; nt < 4; ++nt)
      acc[nt] = __builtin_amdgcn_wmma_f32_16x16x32_bf16(
          false, afrag, false, bfrag[nt], (short)0, acc[nt], false, false);
  };
  for (int ki = 0; ki < NK; ki += 2) {  // static ping-pong (no cndmask)
    kstep(ki + 0, 0, 1);
    kstep(ki + 1, 1, 0);
  }

#pragma unroll
  for (int nt = 0; nt < 4; ++nt) {
#pragma unroll
    for (int r = 0; r < 8; ++r) {
      const int m = tileM + r + 8 * hl;
      const int n = tileN + nt * 16 + l16;
      const float v = acc[nt][r] + bias[n];
      if (mode == 0) {
        outb[(size_t)m * N + n] = (bf16)v;
      } else {
        const float t = v > 0.f ? v : 0.f;
        outf[(size_t)m * N + n] = resid[(size_t)m * N + n] + t;
      }
    }
  }
}

// ---------------------------------------------------------------------------
// Flash attention: block = (b, h, 64 q rows); 4 waves x 16 rows.
// K tiles (32x64, row-major) staged by TDM; V tiles transposed manually.
// Double-buffered (static ping-pong); fragment loads hoisted before WMMAs.
// O = Q + softmax(Q Kt / sqrt(H)) V   (f32 output, residual fused)
// ---------------------------------------------------------------------------
__global__ __launch_bounds__(128) void attn_wmma_kernel(
    const bf16* __restrict__ pQ, const bf16* __restrict__ pK,
    const bf16* __restrict__ pV, const float* __restrict__ Qres,
    float* __restrict__ O) {
  __shared__ bf16 Kt[2][32 * 64];       // [buf][kv_local][d]   (TDM target)
  __shared__ bf16 Vt[2][64 * 32];       // [buf][d][kv_local]   (transposed)
  __shared__ bf16 Pb[4][16 * 32];       // per-wave probability tile

  const int lane = threadIdx.x & 31;
  const int wave = threadIdx.x >> 5;
  const int hl   = lane >> 4;
  const int l16  = lane & 15;
  const int selA = hl ? 16 : 0;
  const bool tdmw = (wave == 0);

  const int b  = blockIdx.z;
  const int h  = blockIdx.y;
  const int q0 = blockIdx.x * 64 + wave * 16;

  const float scale = 0.044194173824159216f;  // 1/sqrt(512)

  // Q A-fragments (head dim 64 = 2 K-chunks of 32)
  const bf16* qrow = pQ + ((size_t)(b * SS + q0 + l16) * HH) + h * DSS;
  v16bf aQ[2];
#pragma unroll
  for (int c = 0; c < 2; ++c) {
    v8bf lo = *(const v8bf*)(qrow + c * 32 + selA);
    v8bf hi = *(const v8bf*)(qrow + c * 32 + selA + 8);
    aQ[c] = combine16(lo, hi);
  }

  // manual transposed staging of one V tile into Vt[buf]
  auto stage_v = [&](int buf, int kv0) {
    const int e     = threadIdx.x * 16;   // 128 threads * 16 elems = 2048
    const int kvloc = e >> 6;
    const int d     = e & 63;
    const size_t gb = ((size_t)(b * SKK + kv0 + kvloc) * HH) + h * DSS + d;
    v8bf v0 = *(const v8bf*)(pV + gb);
    v8bf v1 = *(const v8bf*)(pV + gb + 8);
#pragma unroll
    for (int j = 0; j < 8; ++j) {
      Vt[buf][(d + j) * 32 + kvloc]     = v0[j];
      Vt[buf][(d + 8 + j) * 32 + kvloc] = v1[j];
    }
  };

  // preload tile 0
  if (tdmw)
    tdm_load_tile_bf16(lds_offset_of(&Kt[0][0]),
                       pK + ((size_t)(b * SKK) * HH) + h * DSS, DSS, 32, HH);
  stage_v(0, 0);

  float mr[8], lr[8];
  v8f o[4] = {};
#pragma unroll
  for (int r = 0; r < 8; ++r) { mr[r] = -1e30f; lr[r] = 0.f; }

  const int NT = SKK / 32;              // 64 (even)
  auto kvstep = [&](int it, int cur, int nxt) {
    const int kv0 = it * 32;
    if (tdmw) __builtin_amdgcn_s_wait_tensorcnt(0);
    __syncthreads();                    // Kt/Vt[cur] ready; old readers done
    if (it + 1 < NT) {                  // move tile i+1 under tile i compute
      if (tdmw)
        tdm_load_tile_bf16(
            lds_offset_of(&Kt[nxt][0]),
            pK + ((size_t)(b * SKK + kv0 + 32) * HH) + h * DSS, DSS, 32, HH);
      stage_v(nxt, kv0 + 32);
    }

    // ---- scores: hoist all 4 K fragments, then 4 chained WMMAs ----
    v16bf bk[2][2];
#pragma unroll
    for (int j = 0; j < 2; ++j)
#pragma unroll
      for (int c = 0; c < 2; ++c) {
        const bf16* kp = &Kt[cur][(j * 16 + l16) * 64 + c * 32 + hl * 16];
        bk[j][c] = combine16(*(const v8bf*)kp, *(const v8bf*)(kp + 8));
      }
    v8f s[2] = {};
#pragma unroll
    for (int j = 0; j < 2; ++j)
#pragma unroll
      for (int c = 0; c < 2; ++c)
        s[j] = __builtin_amdgcn_wmma_f32_16x16x32_bf16(
            false, aQ[c], false, bk[j][c], (short)0, s[j], false, false);

    // ---- online softmax across 32 columns (per-row, shfl within half) ----
#pragma unroll
    for (int r = 0; r < 8; ++r) {
      const float s0 = s[0][r] * scale;
      const float s1 = s[1][r] * scale;
      float t = fmaxf(s0, s1);
#pragma unroll
      for (int mk = 1; mk < 16; mk <<= 1) t = fmaxf(t, __shfl_xor(t, mk, 32));
      const float mn    = fmaxf(mr[r], t);
      const float alpha = __expf(mr[r] - mn);
      const float p0 = __expf(s0 - mn);
      const float p1 = __expf(s1 - mn);
      float rs = p0 + p1;
#pragma unroll
      for (int mk = 1; mk < 16; mk <<= 1) rs += __shfl_xor(rs, mk, 32);
      lr[r] = lr[r] * alpha + rs;
      mr[r] = mn;
#pragma unroll
      for (int nt = 0; nt < 4; ++nt) o[nt][r] *= alpha;
      const int prow = r + 8 * hl;
      Pb[wave][prow * 32 + l16]      = (bf16)p0;
      Pb[wave][prow * 32 + 16 + l16] = (bf16)p1;
    }
    __syncthreads();   // P C-layout -> A-layout via LDS

    // ---- O += P(16x32) x V(32x64): hoist P + 4 V fragments ----
    const bf16* pp = &Pb[wave][l16 * 32 + selA];
    v16bf aP = combine16(*(const v8bf*)pp, *(const v8bf*)(pp + 8));
    v16bf bv[4];
#pragma unroll
    for (int nt = 0; nt < 4; ++nt) {
      const bf16* vp = &Vt[cur][(nt * 16 + l16) * 32 + hl * 16];
      bv[nt] = combine16(*(const v8bf*)vp, *(const v8bf*)(vp + 8));
    }
#pragma unroll
    for (int nt = 0; nt < 4; ++nt)
      o[nt] = __builtin_amdgcn_wmma_f32_16x16x32_bf16(
          false, aP, false, bv[nt], (short)0, o[nt], false, false);
  };
  for (int it = 0; it < NT; it += 2) {  // static ping-pong
    kvstep(it + 0, 0, 1);
    kvstep(it + 1, 1, 0);
  }

  // ---- epilogue: normalize by l, add residual Q, f32 out ----
#pragma unroll
  for (int nt = 0; nt < 4; ++nt) {
#pragma unroll
    for (int r = 0; r < 8; ++r) {
      const int q = q0 + r + 8 * hl;
      const int d = nt * 16 + l16;
      const size_t idx = ((size_t)(b * SS + q) * HH) + h * DSS + d;
      O[idx] = o[nt][r] / lr[r] + Qres[idx];
    }
  }
}

// ---------------------------------------------------------------------------
// Launcher
// ---------------------------------------------------------------------------
extern "C" void kernel_launch(void* const* d_in, const int* in_sizes, int n_in,
                              void* d_out, int out_size, void* d_ws,
                              size_t ws_size, hipStream_t stream) {
  const float* Q     = (const float*)d_in[0];
  const float* K     = (const float*)d_in[1];
  const float* wq    = (const float*)d_in[2];
  const float* bq    = (const float*)d_in[3];
  const float* wk    = (const float*)d_in[4];
  const float* bk    = (const float*)d_in[5];
  const float* wv    = (const float*)d_in[6];
  const float* bv    = (const float*)d_in[7];
  const float* wc    = (const float*)d_in[8];
  const float* bc    = (const float*)d_in[9];
  const float* gq    = (const float*)d_in[10];
  const float* betaq = (const float*)d_in[11];
  const float* gk    = (const float*)d_in[12];
  const float* betak = (const float*)d_in[13];
  const float* g0    = (const float*)d_in[14];
  const float* beta0 = (const float*)d_in[15];
  float* out = (float*)d_out;

  char* ws = (char*)d_ws;
  const size_t SZ_ACT = (size_t)MQ * HH * 2;  // 8 MB bf16 activation
  bf16* Qn  = (bf16*)(ws);
  bf16* Kn  = (bf16*)(ws + 1 * SZ_ACT);
  bf16* pQ  = (bf16*)(ws + 2 * SZ_ACT);
  bf16* pK  = (bf16*)(ws + 3 * SZ_ACT);
  bf16* pV  = (bf16*)(ws + 4 * SZ_ACT);
  float* O  = (float*)(ws + 5 * SZ_ACT);                 // 16 MB f32
  bf16* On  = (bf16*)(ws + 5 * SZ_ACT + (size_t)MQ * HH * 4);
  char* wsw = ws + 6 * SZ_ACT + (size_t)MQ * HH * 4;
  const size_t SZ_W = (size_t)HH * HH * 2;               // 512 KB bf16 weight
  bf16* wqb = (bf16*)(wsw);
  bf16* wkb = (bf16*)(wsw + 1 * SZ_W);
  bf16* wvb = (bf16*)(wsw + 2 * SZ_W);
  bf16* wcb = (bf16*)(wsw + 3 * SZ_W);
  float* stats = (float*)(wsw + 4 * SZ_W);   // statsQ[8] statsK[8] statsO[8]
  float* statsQ = stats, *statsK = stats + 8, *statsO = stats + 16;
  float* part = stats + 24;                  // 3 * 512 floats of partials
  float* partQ = part, *partK = part + 512, *partO = part + 1024;

  // --- set-norm stats (deterministic two-stage) ---
  stats_partial_kernel<<<dim3(64, BB), 256, 0, stream>>>(Q, partQ, NPB);
  stats_final_kernel<<<BB, 64, 0, stream>>>(partQ, statsQ);
  stats_partial_kernel<<<dim3(64, BB), 256, 0, stream>>>(K, partK, NPB);
  stats_final_kernel<<<BB, 64, 0, stream>>>(partK, statsK);

  // --- normalize + affine + bf16 convert ---
  norm_cvt_kernel<<<2048, 256, 0, stream>>>(Q, statsQ, gq, betaq, Qn, NPB);
  norm_cvt_kernel<<<2048, 256, 0, stream>>>(K, statsK, gk, betak, Kn, NPB);

  // --- weights to bf16 ---
  cvt_bf16_kernel<<<512, 256, 0, stream>>>(wq, wqb, HH * HH);
  cvt_bf16_kernel<<<512, 256, 0, stream>>>(wk, wkb, HH * HH);
  cvt_bf16_kernel<<<512, 256, 0, stream>>>(wv, wvb, HH * HH);
  cvt_bf16_kernel<<<512, 256, 0, stream>>>(wc, wcb, HH * HH);

  // --- projections (WMMA bf16, TDM-staged weights) ---
  dim3 ggrid(MQ / 128, HH / 64);
  gemm_wmma_kernel<<<ggrid, 256, 0, stream>>>(Qn, wqb, bq, pQ, nullptr,
                                              nullptr, MQ, HH, HH, 0);
  gemm_wmma_kernel<<<ggrid, 256, 0, stream>>>(Kn, wkb, bk, pK, nullptr,
                                              nullptr, MQ, HH, HH, 0);
  gemm_wmma_kernel<<<ggrid, 256, 0, stream>>>(Kn, wvb, bv, pV, nullptr,
                                              nullptr, MQ, HH, HH, 0);

  // --- flash attention + residual (TDM-staged K tiles) ---
  attn_wmma_kernel<<<dim3(SS / 64, NHH, BB), 128, 0, stream>>>(pQ, pK, pV, Q,
                                                               O);

  // --- final set-norm + output projection + relu + residual ---
  stats_partial_kernel<<<dim3(64, BB), 256, 0, stream>>>(O, partO, NPB);
  stats_final_kernel<<<BB, 64, 0, stream>>>(partO, statsO);
  norm_cvt_kernel<<<2048, 256, 0, stream>>>(O, statsO, g0, beta0, On, NPB);
  gemm_wmma_kernel<<<ggrid, 256, 0, stream>>>(On, wcb, bc, nullptr, O, out,
                                              MQ, HH, HH, 1);
}
